// SubdNet_74294344286575
// MI455X (gfx1250) — matrix operand order repaired and checked by reading, not compiled, not run on hardware
//
#include <hip/hip_runtime.h>

// ---------------------------------------------------------------------------
// SubdNet on MI455X (gfx1250): MLP via v_wmma_f32_16x16x32_bf16, weights in LDS
// Operand roles: D = W(A) x X(B) so accumulator rows pack to one b128 store.
// ---------------------------------------------------------------------------

typedef __attribute__((ext_vector_type(16))) __bf16        v16bf;
typedef __attribute__((ext_vector_type(8)))  __bf16        v8bf;
typedef __attribute__((ext_vector_type(8)))  float         v8f;
typedef __attribute__((ext_vector_type(4)))  float         v4f;
typedef __attribute__((ext_vector_type(4)))  unsigned int  v4u;

// LDS layout (dynamic shared memory)
#define SM_WI        0          // 128x128 bf16 [N][K] = 32768 B
#define SM_WH        32768      // 128x128 bf16        = 32768 B
#define SM_WO        65536      // 32x128 bf16         =  8192 B
#define SM_BI        73728      // 128 f32
#define SM_BH        74240      // 128 f32
#define SM_BO        74752      // 32 f32 (pad to 128 B)
#define SM_WAVEBASE  74880
#define PW_ACT       0          // 16x128 bf16 = 4096 B
#define PW_ACT2      4096       // 16x128 bf16 = 4096 B (reused as 16x32 f32 hbuf)
#define PW_LF        8192       // 16x9 f32    =  576 B
#define PER_WAVE     8960
#define SMEM_BYTES   (SM_WAVEBASE + 8 * PER_WAVE)   // 146560 B < 320KB/WGP

// --------------------------- small device helpers --------------------------

__device__ __forceinline__ unsigned short f2bf(float x) {
    unsigned int u = __float_as_uint(x);
    u += 0x7fffu + ((u >> 16) & 1u);          // round to nearest even
    return (unsigned short)(u >> 16);
}
__device__ __forceinline__ unsigned int pack2bf(float lo, float hi) {
    return (unsigned int)f2bf(lo) | ((unsigned int)f2bf(hi) << 16);
}

__device__ __forceinline__ void wave_lds_fence() {
    // LDS ops of a wave are processed in order; this stops the compiler from
    // reordering ds ops across the producer->consumer boundary and drains them.
    __builtin_amdgcn_wave_barrier();
    asm volatile("s_wait_dscnt 0x0" ::: "memory");
    __builtin_amdgcn_wave_barrier();
}

struct F3 { float x, y, z; };
__device__ __forceinline__ F3 f3sub(F3 a, F3 b) { return {a.x-b.x, a.y-b.y, a.z-b.z}; }
__device__ __forceinline__ F3 f3add(F3 a, F3 b) { return {a.x+b.x, a.y+b.y, a.z+b.z}; }
__device__ __forceinline__ F3 f3cross(F3 a, F3 b) {
    return {a.y*b.z - a.z*b.y, a.z*b.x - a.x*b.z, a.x*b.y - a.y*b.x};
}
__device__ __forceinline__ float f3dot(F3 a, F3 b) { return a.x*b.x + a.y*b.y + a.z*b.z; }
__device__ __forceinline__ F3 f3unit(F3 a) {
    float r = rsqrtf(a.x*a.x + a.y*a.y + a.z*a.z);
    return {a.x*r, a.y*r, a.z*r};
}

// A fragment (16x32) = weight tile: rows M = output features f0..f0+15,
// from LDS [NF][128] bf16 (K contiguous).
// ISA A layout: lanes 0-15 M=lane, K={k0..k0+7, k0+16..k0+23};
//               lanes 16-31 M=lane-16, K={k0+8..k0+15, k0+24..k0+31}.
__device__ __forceinline__ v16bf load_w_frag(const unsigned short* w, int lane,
                                             int f0, int k0) {
    int m  = lane & 15;
    int kh = (lane >> 4) * 8;
    const __bf16* p = (const __bf16*)(w + (f0 + m) * 128 + k0 + kh);
    v8bf lo = *(const v8bf*)p;
    v8bf hi = *(const v8bf*)(p + 16);
    return __builtin_shufflevector(lo, hi, 0,1,2,3,4,5,6,7,8,9,10,11,12,13,14,15);
}

// B fragment (32x16) = activation tile: cols N = rows of the MLP batch,
// from LDS [16][128] bf16 (K contiguous).
// ISA B layout: lanes 0-15 N=lane, K={k0..k0+7, k0+16..k0+23}; lanes 16-31 N=lane-16, K={k0+8..k0+15, k0+24..k0+31}.
__device__ __forceinline__ v16bf load_x_frag(const unsigned short* x, int lane, int k0) {
    int n  = lane & 15;
    int kb = k0 + ((lane >> 4) << 3);
    const __bf16* p = (const __bf16*)(x + n * 128 + kb);
    v8bf lo = *(const v8bf*)p;
    v8bf hi = *(const v8bf*)(p + 16);
    return __builtin_shufflevector(lo, hi, 0,1,2,3,4,5,6,7,8,9,10,11,12,13,14,15);
}

// Hidden layer: [16 rows x 128] bf16 -> [16 x 128] bf16 with fused ReLU.
// D=(W)x(X): per lane the 8 accumulators are contiguous output features of one
// row -> ReLU + pack -> single ds_store_b128.
__device__ __forceinline__ void mlp_layer128(const unsigned short* xLDS,
                                             const unsigned short* wLDS,
                                             const float* biasLDS,
                                             unsigned short* oLDS, int lane) {
    int nl    = lane & 15;            // row (N)
    int mbase = (lane >> 4) * 8;      // feature sub-block (M)
    v16bf b0 = load_x_frag(xLDS, lane, 0);
    v16bf b1 = load_x_frag(xLDS, lane, 32);
    v16bf b2 = load_x_frag(xLDS, lane, 64);
    v16bf b3 = load_x_frag(xLDS, lane, 96);
    for (int ot = 0; ot < 8; ++ot) {
        int f0 = ot * 16;
        v8f c = *(const v8f*)(biasLDS + f0 + mbase);
        c = __builtin_amdgcn_wmma_f32_16x16x32_bf16(false, load_w_frag(wLDS, lane, f0, 0),
                                                    false, b0, (short)0, c, false, false);
        c = __builtin_amdgcn_wmma_f32_16x16x32_bf16(false, load_w_frag(wLDS, lane, f0, 32),
                                                    false, b1, (short)0, c, false, false);
        c = __builtin_amdgcn_wmma_f32_16x16x32_bf16(false, load_w_frag(wLDS, lane, f0, 64),
                                                    false, b2, (short)0, c, false, false);
        c = __builtin_amdgcn_wmma_f32_16x16x32_bf16(false, load_w_frag(wLDS, lane, f0, 96),
                                                    false, b3, (short)0, c, false, false);
        v4u o;
        o[0] = pack2bf(fmaxf(c[0], 0.0f), fmaxf(c[1], 0.0f));
        o[1] = pack2bf(fmaxf(c[2], 0.0f), fmaxf(c[3], 0.0f));
        o[2] = pack2bf(fmaxf(c[4], 0.0f), fmaxf(c[5], 0.0f));
        o[3] = pack2bf(fmaxf(c[6], 0.0f), fmaxf(c[7], 0.0f));
        *(v4u*)(oLDS + nl * 128 + f0 + mbase) = o;      // 16 B aligned
    }
}

// --------------------------- main fused kernel -----------------------------
// One wave = one tile of 16 half-flaps. Gather -> flap_normalize -> 3-layer
// MLP (WMMA) -> local2global -> (mode 0) atomic one-ring accumulate into
// sum/cnt, or (mode 1) store per-half-flap features to fhf scratch.
__global__ void __launch_bounds__(256)
flap_mlp_kernel(const float* __restrict__ fv,
                const int*   __restrict__ hfIdx,
                const unsigned short* __restrict__ wiT,   // [128][128] bf16, [N][K]
                const unsigned short* __restrict__ whT,   // [128][128] bf16
                const unsigned short* __restrict__ woT,   // [32][128] bf16
                const float* __restrict__ bi,
                const float* __restrict__ bh,
                const float* __restrict__ bo,
                float* __restrict__ dstA,                 // mode0: sum[nV][32]; mode1: fhf[H][32]
                float* __restrict__ cnt,                  // mode0 only
                int nTiles, int mode) {
    extern __shared__ __align__(16) char smem[];
    int tid = threadIdx.x;

    // ---- cooperative weight + bias load into LDS (block-wide) ----
    {
        unsigned int* d;
        const unsigned int* g;
        d = (unsigned int*)(smem + SM_WI); g = (const unsigned int*)wiT;
        for (int i = tid; i < 8192; i += 256) d[i] = g[i];
        d = (unsigned int*)(smem + SM_WH); g = (const unsigned int*)whT;
        for (int i = tid; i < 8192; i += 256) d[i] = g[i];
        d = (unsigned int*)(smem + SM_WO); g = (const unsigned int*)woT;
        for (int i = tid; i < 2048; i += 256) d[i] = g[i];
        float* fb;
        fb = (float*)(smem + SM_BI); for (int i = tid; i < 128; i += 256) fb[i] = bi[i];
        fb = (float*)(smem + SM_BH); for (int i = tid; i < 128; i += 256) fb[i] = bh[i];
        fb = (float*)(smem + SM_BO); for (int i = tid; i < 32;  i += 256) fb[i] = bo[i];
    }
    __syncthreads();

    int wave = tid >> 5;
    int lane = tid & 31;
    int tile = blockIdx.x * 8 + wave;
    if (tile >= nTiles) return;                 // whole-wave exit, no more barriers

    char* pw = smem + SM_WAVEBASE + wave * PER_WAVE;
    unsigned short* act  = (unsigned short*)(pw + PW_ACT);   // [16][128] bf16
    unsigned short* act2 = (unsigned short*)(pw + PW_ACT2);  // [16][128] bf16
    float*          sLF  = (float*)(pw + PW_LF);             // [16][9]
    const unsigned short* swi = (const unsigned short*)(smem + SM_WI);
    const unsigned short* swh = (const unsigned short*)(smem + SM_WH);
    const unsigned short* swo = (const unsigned short*)(smem + SM_WO);
    const float* sbi = (const float*)(smem + SM_BI);
    const float* sbh = (const float*)(smem + SM_BH);
    const float* sbo = (const float*)(smem + SM_BO);

    int flap0 = tile * 16;

    // ---- gather: copy whole 32-channel rows for all (row, vert) pairs ----
    // 16 rows x 4 verts = 64 pairs; each lane handles 2 pairs, b128 loads/stores.
    #pragma unroll
    for (int pp = 0; pp < 2; ++pp) {
        int p   = lane + pp * 32;
        int row = p >> 2;
        int v   = p & 3;
        int g   = hfIdx[4 * (flap0 + row) + v];
        const float* src = fv + (size_t)g * 32;
        unsigned short* dst = act + row * 128 + v * 32;
        #pragma unroll
        for (int q = 0; q < 4; ++q) {
            v4f f0 = *(const v4f*)(src + q * 8);
            v4f f1 = *(const v4f*)(src + q * 8 + 4);
            v4u o;
            o[0] = pack2bf(f0[0], f0[1]);
            o[1] = pack2bf(f0[2], f0[3]);
            o[2] = pack2bf(f1[0], f1[1]);
            o[3] = pack2bf(f1[2], f1[3]);
            *(v4u*)(dst + q * 8) = o;
        }
    }
    wave_lds_fence();

    // ---- flap frame + local positions (one lane per half-flap) ----
    // Overwrites the 12 position slots with frame-local coords (v0 -> 0).
    if (lane < 16) {
        int f  = flap0 + lane;
        int i0 = hfIdx[4 * f + 0];
        int i1 = hfIdx[4 * f + 1];
        int i2 = hfIdx[4 * f + 2];
        int i3 = hfIdx[4 * f + 3];
        F3 v0 = {fv[i0 * 32 + 0], fv[i0 * 32 + 1], fv[i0 * 32 + 2]};
        F3 v1 = {fv[i1 * 32 + 0], fv[i1 * 32 + 1], fv[i1 * 32 + 2]};
        F3 v2 = {fv[i2 * 32 + 0], fv[i2 * 32 + 1], fv[i2 * 32 + 2]};
        F3 v3 = {fv[i3 * 32 + 0], fv[i3 * 32 + 1], fv[i3 * 32 + 2]};
        F3 e1  = f3sub(v1, v0);
        F3 b1  = f3unit(e1);
        F3 fn0 = f3unit(f3cross(e1, f3sub(v2, v0)));
        F3 fn1 = f3unit(f3cross(f3sub(v0, v1), f3sub(v3, v1)));
        F3 b3  = f3unit(f3add(fn0, fn1));
        F3 b2  = f3cross(b3, b1);
        sLF[lane * 9 + 0] = b1.x; sLF[lane * 9 + 1] = b1.y; sLF[lane * 9 + 2] = b1.z;
        sLF[lane * 9 + 3] = b2.x; sLF[lane * 9 + 4] = b2.y; sLF[lane * 9 + 5] = b2.z;
        sLF[lane * 9 + 6] = b3.x; sLF[lane * 9 + 7] = b3.y; sLF[lane * 9 + 8] = b3.z;
        unsigned short* ar = act + lane * 128;
        *(unsigned int*)(ar + 0) = 0u;            // k=0,1 zero pad
        ar[2] = 0;                                // k=2 zero pad
        F3 dv;
        dv = f3sub(v1, v0);
        *(unsigned int*)(ar + 32) = pack2bf(f3dot(dv, b1), f3dot(dv, b2));
        ar[34] = f2bf(f3dot(dv, b3));
        dv = f3sub(v2, v0);
        *(unsigned int*)(ar + 64) = pack2bf(f3dot(dv, b1), f3dot(dv, b2));
        ar[66] = f2bf(f3dot(dv, b3));
        dv = f3sub(v3, v0);
        *(unsigned int*)(ar + 96) = pack2bf(f3dot(dv, b1), f3dot(dv, b2));
        ar[98] = f2bf(f3dot(dv, b3));
    }
    wave_lds_fence();

    // ---- layer 1: Wi x act -> act2, ReLU ----
    mlp_layer128(act, swi, sbi, act2, lane);
    wave_lds_fence();

    // ---- layer 2: Wh x act2 -> act, ReLU ----
    mlp_layer128(act2, swh, sbh, act, lane);
    wave_lds_fence();

    // ---- layer 3: Wo x act -> f32 hbuf[16][32] (no ReLU) ----
    float* hbuf = (float*)act2;                    // reuse: 16*32 f32 = 2048 B
    {
        int nl    = lane & 15;
        int mbase = (lane >> 4) * 8;
        v16bf b0 = load_x_frag(act, lane, 0);
        v16bf b1 = load_x_frag(act, lane, 32);
        v16bf b2 = load_x_frag(act, lane, 64);
        v16bf b3 = load_x_frag(act, lane, 96);
        #pragma unroll
        for (int ot = 0; ot < 2; ++ot) {
            int f0 = ot * 16;
            v8f c = *(const v8f*)(sbo + f0 + mbase);
            c = __builtin_amdgcn_wmma_f32_16x16x32_bf16(false, load_w_frag(swo, lane, f0, 0),
                                                        false, b0, (short)0, c, false, false);
            c = __builtin_amdgcn_wmma_f32_16x16x32_bf16(false, load_w_frag(swo, lane, f0, 32),
                                                        false, b1, (short)0, c, false, false);
            c = __builtin_amdgcn_wmma_f32_16x16x32_bf16(false, load_w_frag(swo, lane, f0, 64),
                                                        false, b2, (short)0, c, false, false);
            c = __builtin_amdgcn_wmma_f32_16x16x32_bf16(false, load_w_frag(swo, lane, f0, 96),
                                                        false, b3, (short)0, c, false, false);
            v4f s0 = {c[0], c[1], c[2], c[3]};
            v4f s1 = {c[4], c[5], c[6], c[7]};
            *(v4f*)(hbuf + nl * 32 + f0 + mbase)     = s0;
            *(v4f*)(hbuf + nl * 32 + f0 + mbase + 4) = s1;
        }
    }
    wave_lds_fence();

    // ---- local2global + output (2 lanes per row, 16 channels each) ----
    {
        int row  = lane >> 1;
        int half = lane & 1;
        int c0   = half * 16;
        float h0 = hbuf[row * 32 + 0];
        float h1 = hbuf[row * 32 + 1];
        float h2 = hbuf[row * 32 + 2];
        const float* LFr = sLF + row * 9;
        if (mode == 0) {
            int seg = hfIdx[4 * (flap0 + row) + 0];
            float* dst = dstA + (size_t)seg * 32;
            #pragma unroll
            for (int j = 0; j < 16; ++j) {
                int c = c0 + j;
                float val = (c < 3)
                          ? (h0 * LFr[c] + h1 * LFr[3 + c] + h2 * LFr[6 + c])
                          : hbuf[row * 32 + c];
                atomicAdd(dst + c, val);
            }
            if (half == 0) atomicAdd(cnt + seg, 1.0f);
        } else {
            float* dst = dstA + (size_t)(flap0 + row) * 32;
            #pragma unroll
            for (int j = 0; j < 16; ++j) {
                int c = c0 + j;
                float val = (c < 3)
                          ? (h0 * LFr[c] + h1 * LFr[3 + c] + h2 * LFr[6 + c])
                          : hbuf[row * 32 + c];
                dst[c] = val;
            }
        }
    }
}

// --------------------------- auxiliary kernels -----------------------------

__global__ void cvt_wi_kernel(const float* __restrict__ W_in, unsigned short* __restrict__ wiT) {
    int i = blockIdx.x * blockDim.x + threadIdx.x;     // 4*128*128
    if (i >= 4 * 128 * 128) return;
    int s = i >> 14, n = (i >> 7) & 127, k = i & 127;
    float v = (k < 3) ? 0.0f : W_in[(s * 125 + (k - 3)) * 128 + n];
    wiT[i] = f2bf(v);
}
__global__ void cvt_wh_kernel(const float* __restrict__ W_h, unsigned short* __restrict__ whT) {
    int i = blockIdx.x * blockDim.x + threadIdx.x;     // 4*128*128
    if (i >= 4 * 128 * 128) return;
    int s = i >> 14, n = (i >> 7) & 127, k = i & 127;
    whT[i] = f2bf(W_h[(s * 128 + k) * 128 + n]);
}
__global__ void cvt_wo_kernel(const float* __restrict__ W_out, unsigned short* __restrict__ woT) {
    int i = blockIdx.x * blockDim.x + threadIdx.x;     // 4*32*128
    if (i >= 4 * 32 * 128) return;
    int s = i >> 12, n = (i >> 7) & 31, k = i & 127;
    woT[i] = f2bf(W_out[(s * 128 + k) * 32 + n]);
}
__global__ void zero_kernel(float* __restrict__ p, int n) {
    int i = blockIdx.x * blockDim.x + threadIdx.x;
    if (i < n) p[i] = 0.0f;
}
__global__ void finalize_v_kernel(const float* __restrict__ sum, const float* __restrict__ cnt,
                                  const float* __restrict__ fv_in, float* __restrict__ fv_out,
                                  int nV) {
    int i = blockIdx.x * blockDim.x + threadIdx.x;
    if (i >= nV * 32) return;
    int v = i >> 5, c = i & 31;
    float d = cnt[v];
    d = (d > 0.0f) ? d : 1.0f;
    float val = sum[i] / d;
    if (c < 3) val += fv_in[v * 32 + c];
    fv_out[i] = val;
}
__global__ void ve_kernel(const float* __restrict__ fv, const int* __restrict__ hfIdx,
                          float* __restrict__ Ve, int nE) {
    int i = blockIdx.x * blockDim.x + threadIdx.x;
    if (i >= nE * 3) return;
    int e = i / 3, j = i - e * 3;
    int a = hfIdx[8 * e + 0], b = hfIdx[8 * e + 1];
    int c = hfIdx[8 * e + 4], d = hfIdx[8 * e + 5];
    Ve[i] = 0.25f * (fv[a * 32 + j] + fv[b * 32 + j] + fv[c * 32 + j] + fv[d * 32 + j]);
}
__global__ void finalize_e_kernel(const float* __restrict__ fhf, const float* __restrict__ Ve,
                                  float* __restrict__ fv_odd, int nE) {
    int i = blockIdx.x * blockDim.x + threadIdx.x;
    if (i >= nE * 32) return;
    int e = i >> 5, c = i & 31;
    float val = 0.5f * (fhf[(size_t)(2 * e) * 32 + c] + fhf[(size_t)(2 * e + 1) * 32 + c]);
    if (c < 3) val += Ve[e * 3 + c];
    fv_odd[i] = val;
}
__global__ void copy_pos_kernel(const float* __restrict__ fv, float* __restrict__ out, int n) {
    int i = blockIdx.x * blockDim.x + threadIdx.x;
    if (i >= n * 3) return;
    int v = i / 3, j = i - v * 3;
    out[i] = fv[v * 32 + j];
}

// ------------------------------- launcher ----------------------------------

extern "C" void kernel_launch(void* const* d_in, const int* in_sizes, int n_in,
                              void* d_out, int out_size, void* d_ws, size_t ws_size,
                              hipStream_t stream) {
    const float* fv0   = (const float*)d_in[0];
    const float* W_in  = (const float*)d_in[1];
    const float* b_in  = (const float*)d_in[2];
    const float* W_h   = (const float*)d_in[3];
    const float* b_h   = (const float*)d_in[4];
    const float* W_out = (const float*)d_in[5];
    const float* b_out = (const float*)d_in[6];
    const int*   hf0   = (const int*)d_in[7];
    const int*   hf1   = (const int*)d_in[8];

    // workspace carve-out (256 B aligned)
    char* ws = (char*)d_ws;
    size_t off = 0;
    auto carve = [&](size_t bytes) -> char* {
        char* p = ws + off;
        off = (off + bytes + 255) & ~(size_t)255;
        return p;
    };
    float* fv1 = (float*)carve((size_t)120000 * 32 * 4);
    float* fv2 = (float*)carve((size_t)390000 * 32 * 4);
    float* sum = (float*)carve((size_t)120000 * 32 * 4);
    float* cnt = (float*)carve((size_t)120000 * 4);
    float* Ve  = (float*)carve((size_t)270000 * 3 * 4);
    float* fhf = (float*)carve((size_t)540000 * 32 * 4);
    unsigned short* wiT = (unsigned short*)carve((size_t)4 * 128 * 128 * 2);
    unsigned short* whT = (unsigned short*)carve((size_t)4 * 128 * 128 * 2);
    unsigned short* woT = (unsigned short*)carve((size_t)4 * 32 * 128 * 2);

    hipFuncSetAttribute((const void*)flap_mlp_kernel,
                        hipFuncAttributeMaxDynamicSharedMemorySize, SMEM_BYTES);

    // bf16 weight conversion (deterministic, tiny)
    cvt_wi_kernel<<<(4 * 128 * 128 + 255) / 256, 256, 0, stream>>>(W_in, wiT);
    cvt_wh_kernel<<<(4 * 128 * 128 + 255) / 256, 256, 0, stream>>>(W_h, whT);
    cvt_wo_kernel<<<(4 * 32 * 128 + 255) / 256, 256, 0, stream>>>(W_out, woT);

    struct Lv { int nV, nE; const int* hf; const float* fin; float* fout; };
    Lv lvs[2] = { {30000,  90000,  hf0, fv0, fv1},
                  {120000, 270000, hf1, fv1, fv2} };

    for (int ii = 0; ii < 2; ++ii) {
        int nV = lvs[ii].nV, nE = lvs[ii].nE;
        int H = 2 * nE;
        int nTiles = H / 16;                        // exact at both levels
        int nBlocks = (nTiles + 7) / 8;
        const int* hf = lvs[ii].hf;
        const float* fin = lvs[ii].fin;
        float* fout = lvs[ii].fout;
        int sv = 2 * ii, se = 2 * ii + 1;

        zero_kernel<<<(nV * 32 + 255) / 256, 256, 0, stream>>>(sum, nV * 32);
        zero_kernel<<<(nV + 255) / 256, 256, 0, stream>>>(cnt, nV);

        // vertex MLP pass: accumulate one-ring sums
        flap_mlp_kernel<<<nBlocks, 256, SMEM_BYTES, stream>>>(
            fin, hf, wiT + sv * 16384, whT + sv * 16384, woT + sv * 4096,
            b_in + sv * 128, b_h + sv * 128, b_out + sv * 32,
            sum, cnt, nTiles, 0);

        finalize_v_kernel<<<(nV * 32 + 255) / 256, 256, 0, stream>>>(sum, cnt, fin, fout, nV);
        ve_kernel<<<(nE * 3 + 255) / 256, 256, 0, stream>>>(fout, hf, Ve, nE);

        // edge MLP pass: store per-half-flap features
        flap_mlp_kernel<<<nBlocks, 256, SMEM_BYTES, stream>>>(
            fout, hf, wiT + se * 16384, whT + se * 16384, woT + se * 4096,
            b_in + se * 128, b_h + se * 128, b_out + se * 32,
            fhf, nullptr, nTiles, 1);

        finalize_e_kernel<<<(nE * 32 + 255) / 256, 256, 0, stream>>>(
            fhf, Ve, fout + (size_t)nV * 32, nE);
    }

    float* out = (float*)d_out;
    copy_pos_kernel<<<(30000 * 3 + 255) / 256, 256, 0, stream>>>(fv0, out, 30000);
    copy_pos_kernel<<<(120000 * 3 + 255) / 256, 256, 0, stream>>>(fv1, out + 90000, 120000);
    copy_pos_kernel<<<(390000 * 3 + 255) / 256, 256, 0, stream>>>(fv2, out + 450000, 390000);
}
